// SSMBlock_6828998000787
// MI455X (gfx1250) — matrix operand order repaired and checked by reading, not compiled, not run on hardware
//
#include <hip/hip_runtime.h>
#include <math.h>

typedef __bf16 bf16;
typedef __attribute__((ext_vector_type(16))) __bf16 v16bf;
typedef __attribute__((ext_vector_type(8)))  __bf16 bf16x8;
typedef __attribute__((ext_vector_type(8)))  float  v8f;

#define B_  4
#define T_  4096
#define D_  1024
#define H_  4096
#define M_  (B_*T_)      // 16384 rows
#define CH_ 64           // scan chunks
#define L_  (T_/CH_)     // 64 steps per chunk

union FragAB { v16bf v; bf16x8 h[2]; };

// Async global -> LDS 128-bit copy (CDNA5, tracked by ASYNCcnt).
#define ASYNC_B128(lds, gptr) \
    asm volatile("global_load_async_to_lds_b128 %0, %1, off" :: "v"(lds), "v"(gptr) : "memory")
#define WAIT_ASYNC_4() asm volatile("s_wait_asynccnt 0x4" ::: "memory")
#define WAIT_ASYNC_0() asm volatile("s_wait_asynccnt 0x0" ::: "memory")

__device__ __forceinline__ unsigned lds_off(const void* p) {
    return (unsigned)(size_t)p;   // low 32 bits of generic addr = LDS byte address
}

// ---------------- elementwise f32 -> bf16 ----------------
__global__ void k_cvt_bf16(const float* __restrict__ in, bf16* __restrict__ out, int n) {
    int i = blockIdx.x * blockDim.x + threadIdx.x;
    if (i < n) out[i] = (bf16)in[i];
}

// ---------------- rmsnorm1: out = x/rms * n1_w * b  (f32) -------------
__global__ void k_rms1(const float* __restrict__ x, const float* __restrict__ nw,
                       const float* __restrict__ bvec, float* __restrict__ outp) {
    int wid = threadIdx.x >> 5, lane = threadIdx.x & 31;
    int row = blockIdx.x * 8 + wid;
    const float* xr = x + (size_t)row * D_;
    float v[32]; float s = 0.f;
    #pragma unroll
    for (int i = 0; i < 32; ++i) { v[i] = xr[lane + i * 32]; s += v[i] * v[i]; }
    #pragma unroll
    for (int m = 16; m; m >>= 1) s += __shfl_xor(s, m, 32);
    float inv = rsqrtf(s * (1.f / D_) + 1e-5f);
    float* orow = outp + (size_t)row * D_;
    #pragma unroll
    for (int i = 0; i < 32; ++i) {
        int d = lane + i * 32;
        orow[d] = v[i] * inv * nw[d] * bvec[d];
    }
}

// ---------------- rmsnorm2: out = x/rms * n2_w  (bf16) -------------
__global__ void k_rms2(const float* __restrict__ x, const float* __restrict__ nw,
                       bf16* __restrict__ outp) {
    int wid = threadIdx.x >> 5, lane = threadIdx.x & 31;
    int row = blockIdx.x * 8 + wid;
    const float* xr = x + (size_t)row * D_;
    float v[32]; float s = 0.f;
    #pragma unroll
    for (int i = 0; i < 32; ++i) { v[i] = xr[lane + i * 32]; s += v[i] * v[i]; }
    #pragma unroll
    for (int m = 16; m; m >>= 1) s += __shfl_xor(s, m, 32);
    float inv = rsqrtf(s * (1.f / D_) + 1e-5f);
    bf16* orow = outp + (size_t)row * D_;
    #pragma unroll
    for (int i = 0; i < 32; ++i) {
        int d = lane + i * 32;
        orow[d] = (bf16)(v[i] * inv * nw[d]);
    }
}

// ---------------- scan pass A: in-place local scans per chunk ----------------
__global__ void k_scan_local(float* __restrict__ bx, float* __restrict__ tail,
                             const float* __restrict__ a) {
    int g = blockIdx.x * blockDim.x + threadIdx.x;   // B*CH*D threads
    int d = g & (D_ - 1);
    int rest = g >> 10;
    int c = rest & (CH_ - 1);
    int b = rest >> 6;
    float at = tanhf(a[d]);
    size_t base = ((size_t)(b * T_ + c * L_)) * D_ + d;
    float prev = 0.f;
    for (int i = 0; i < L_; ++i) {
        float vv = at * prev + bx[base + (size_t)i * D_];
        bx[base + (size_t)i * D_] = vv;
        prev = vv;
    }
    tail[(size_t)(b * CH_ + c) * D_ + d] = prev;
}

// ---------------- scan pass B: sequential carries over chunks ----------------
__global__ void k_scan_carry(const float* __restrict__ tail, float* __restrict__ s0,
                             const float* __restrict__ a) {
    int g = blockIdx.x * blockDim.x + threadIdx.x;   // B*D threads
    int d = g & (D_ - 1);
    int b = g >> 10;
    float at = tanhf(a[d]);
    float atL = at;
    #pragma unroll
    for (int i = 0; i < 6; ++i) atL *= atL;          // at^64
    float carry = 0.f;
    for (int c = 0; c < CH_; ++c) {
        size_t idx = (size_t)(b * CH_ + c) * D_ + d;
        s0[idx] = carry;
        carry = atL * carry + tail[idx];
    }
}

// ---------------- scan pass C: y = at^(i+1)*s0 + local -> bf16 ----------------
__global__ void k_scan_fix(const float* __restrict__ local_, const float* __restrict__ s0,
                           const float* __restrict__ a, bf16* __restrict__ y) {
    int g = blockIdx.x * blockDim.x + threadIdx.x;   // B*CH*D threads
    int d = g & (D_ - 1);
    int rest = g >> 10;
    int c = rest & (CH_ - 1);
    int b = rest >> 6;
    float at = tanhf(a[d]);
    float sv = s0[(size_t)(b * CH_ + c) * D_ + d];
    size_t base = ((size_t)(b * T_ + c * L_)) * D_ + d;
    float p = at;
    for (int i = 0; i < L_; ++i) {
        float vy = p * sv + local_[base + (size_t)i * D_];
        y[base + (size_t)i * D_] = (bf16)vy;
        p *= at;
    }
}

// ---------------- GEMM (resid): out = res + A(bf16)[M,K] x B(bf16)[N,K]^T + bias ----
// block 128x128, 8 waves (4x2), wave tile 32x64, KSTEP=32, pair-unrolled double
// buffering with async global->LDS staging + v_wmma_f32_16x16x32_bf16
__launch_bounds__(256)
__global__ void k_gemm_resid(const bf16* __restrict__ A, const bf16* __restrict__ Bm,
                             const float* __restrict__ bias, const float* __restrict__ res,
                             float* __restrict__ out, int N, int K) {
    constexpr int LDA = 48;
    __shared__ bf16 As[2][128 * LDA];
    __shared__ bf16 Bs[2][128 * LDA];
    int tid = threadIdx.x;
    int wid = tid >> 5, lane = tid & 31;
    int wm = wid >> 1, wn = wid & 1;
    int bm = blockIdx.y * 128, bn = blockIdx.x * 128;

    v8f acc[2][4];
    #pragma unroll
    for (int mi = 0; mi < 2; ++mi)
        #pragma unroll
        for (int ni = 0; ni < 4; ++ni)
            acc[mi][ni] = v8f{0.f, 0.f, 0.f, 0.f, 0.f, 0.f, 0.f, 0.f};

    // staging assignment: 128x32-half tile = 512 b128 vectors; 2 per thread per matrix
    int r0 = tid >> 2,         c0 = (tid & 3) * 8;
    int r1 = (tid + 256) >> 2, c1 = (tid & 3) * 8;
    const bf16* gA0 = A  + (size_t)(bm + r0) * K + c0;
    const bf16* gA1 = A  + (size_t)(bm + r1) * K + c1;
    const bf16* gB0 = Bm + (size_t)(bn + r0) * K + c0;
    const bf16* gB1 = Bm + (size_t)(bn + r1) * K + c1;
    unsigned lA0[2], lA1[2], lB0[2], lB1[2];
    #pragma unroll
    for (int p = 0; p < 2; ++p) {
        lA0[p] = lds_off(&As[p][r0 * LDA + c0]);
        lA1[p] = lds_off(&As[p][r1 * LDA + c1]);
        lB0[p] = lds_off(&Bs[p][r0 * LDA + c0]);
        lB1[p] = lds_off(&Bs[p][r1 * LDA + c1]);
    }

    int lrow = lane & 15;
    int ka = (lane >> 4) * 16;    // A-frag K offset (16 contiguous)
    int kb = (lane >> 4) * 16;    // B-frag K offset (16 contiguous)

    auto issue = [&](int q) {
        ASYNC_B128(lA0[q], gA0); ASYNC_B128(lA1[q], gA1);
        ASYNC_B128(lB0[q], gB0); ASYNC_B128(lB1[q], gB1);
        gA0 += 32; gA1 += 32; gB0 += 32; gB1 += 32;
    };
    auto compute = [&](int p) {
        const bf16* Ab = &As[p][0];
        const bf16* Bb = &Bs[p][0];
        FragAB fa[2], fb[4];
        #pragma unroll
        for (int mi = 0; mi < 2; ++mi) {
            int r = wm * 32 + mi * 16 + lrow;
            fa[mi].h[0] = *(const bf16x8*)&Ab[r * LDA + ka];
            fa[mi].h[1] = *(const bf16x8*)&Ab[r * LDA + ka + 8];
        }
        #pragma unroll
        for (int ni = 0; ni < 4; ++ni) {
            int r = wn * 64 + ni * 16 + lrow;
            fb[ni].h[0] = *(const bf16x8*)&Bb[r * LDA + kb];
            fb[ni].h[1] = *(const bf16x8*)&Bb[r * LDA + kb + 8];
        }
        #pragma unroll
        for (int mi = 0; mi < 2; ++mi)
            #pragma unroll
            for (int ni = 0; ni < 4; ++ni)
                acc[mi][ni] = __builtin_amdgcn_wmma_f32_16x16x32_bf16(
                    false, fa[mi].v, false, fb[ni].v, (short)0, acc[mi][ni], false, false);
    };

    int nk = K / 32;              // even (32 or 128)
    issue(0);
    for (int i = 0; i < nk - 2; i += 2) {
        issue(1); WAIT_ASYNC_4(); __syncthreads(); compute(0); __syncthreads();
        issue(0); WAIT_ASYNC_4(); __syncthreads(); compute(1); __syncthreads();
    }
    issue(1); WAIT_ASYNC_4(); __syncthreads(); compute(0); __syncthreads();
    WAIT_ASYNC_0(); __syncthreads(); compute(1);

    int lm = 8 * (lane >> 4), ln = lane & 15;
    #pragma unroll
    for (int mi = 0; mi < 2; ++mi)
        #pragma unroll
        for (int ni = 0; ni < 4; ++ni) {
            int col = bn + wn * 64 + ni * 16 + ln;
            float bv = bias[col];
            #pragma unroll
            for (int r = 0; r < 8; ++r) {
                int row = bm + wm * 32 + mi * 16 + lm + r;
                size_t idx = (size_t)row * N + col;
                out[idx] = res[idx] + acc[mi][ni][r] + bv;
            }
        }
}

// ---------------- fused dual GEMM + SiLU gate: h = silu(A B2^T + b2) * (A B1^T + b1) ---
// block 128x64, 8 waves (4x2), wave tile 32x32, pair-unrolled async double buffering
__launch_bounds__(256)
__global__ void k_gemm_dual_silu(const bf16* __restrict__ A,
                                 const bf16* __restrict__ B1, const bf16* __restrict__ B2,
                                 const float* __restrict__ bias1, const float* __restrict__ bias2,
                                 bf16* __restrict__ out, int N, int K) {
    constexpr int LDA = 48;
    __shared__ bf16 As[2][128 * LDA];
    __shared__ bf16 Bs1[2][64 * LDA];
    __shared__ bf16 Bs2[2][64 * LDA];
    int tid = threadIdx.x;
    int wid = tid >> 5, lane = tid & 31;
    int wm = wid >> 1, wn = wid & 1;
    int bm = blockIdx.y * 128, bn = blockIdx.x * 64;

    v8f acc1[2][2], acc2[2][2];
    #pragma unroll
    for (int mi = 0; mi < 2; ++mi)
        #pragma unroll
        for (int ni = 0; ni < 2; ++ni) {
            acc1[mi][ni] = v8f{0.f, 0.f, 0.f, 0.f, 0.f, 0.f, 0.f, 0.f};
            acc2[mi][ni] = v8f{0.f, 0.f, 0.f, 0.f, 0.f, 0.f, 0.f, 0.f};
        }

    // staging: A = 2 vecs/thread; B1,B2 = 1 vec/thread each (64x32 halves = 256 vecs)
    int r0 = tid >> 2,         c0 = (tid & 3) * 8;
    int r1 = (tid + 256) >> 2, c1 = (tid & 3) * 8;
    const bf16* gA0 = A  + (size_t)(bm + r0) * K + c0;
    const bf16* gA1 = A  + (size_t)(bm + r1) * K + c1;
    const bf16* gB1 = B1 + (size_t)(bn + r0) * K + c0;
    const bf16* gB2 = B2 + (size_t)(bn + r0) * K + c0;
    unsigned lA0[2], lA1[2], lB1[2], lB2[2];
    #pragma unroll
    for (int p = 0; p < 2; ++p) {
        lA0[p] = lds_off(&As[p][r0 * LDA + c0]);
        lA1[p] = lds_off(&As[p][r1 * LDA + c1]);
        lB1[p] = lds_off(&Bs1[p][r0 * LDA + c0]);
        lB2[p] = lds_off(&Bs2[p][r0 * LDA + c0]);
    }

    int lrow = lane & 15;
    int ka = (lane >> 4) * 16;
    int kb = (lane >> 4) * 16;

    auto issue = [&](int q) {
        ASYNC_B128(lA0[q], gA0); ASYNC_B128(lA1[q], gA1);
        ASYNC_B128(lB1[q], gB1); ASYNC_B128(lB2[q], gB2);
        gA0 += 32; gA1 += 32; gB1 += 32; gB2 += 32;
    };
    auto compute = [&](int p) {
        const bf16* Ab  = &As[p][0];
        const bf16* Bb1 = &Bs1[p][0];
        const bf16* Bb2 = &Bs2[p][0];
        FragAB fa[2], f1[2], f2[2];
        #pragma unroll
        for (int mi = 0; mi < 2; ++mi) {
            int r = wm * 32 + mi * 16 + lrow;
            fa[mi].h[0] = *(const bf16x8*)&Ab[r * LDA + ka];
            fa[mi].h[1] = *(const bf16x8*)&Ab[r * LDA + ka + 8];
        }
        #pragma unroll
        for (int ni = 0; ni < 2; ++ni) {
            int r = wn * 32 + ni * 16 + lrow;
            f1[ni].h[0] = *(const bf16x8*)&Bb1[r * LDA + kb];
            f1[ni].h[1] = *(const bf16x8*)&Bb1[r * LDA + kb + 8];
            f2[ni].h[0] = *(const bf16x8*)&Bb2[r * LDA + kb];
            f2[ni].h[1] = *(const bf16x8*)&Bb2[r * LDA + kb + 8];
        }
        #pragma unroll
        for (int mi = 0; mi < 2; ++mi)
            #pragma unroll
            for (int ni = 0; ni < 2; ++ni) {
                acc1[mi][ni] = __builtin_amdgcn_wmma_f32_16x16x32_bf16(
                    false, fa[mi].v, false, f1[ni].v, (short)0, acc1[mi][ni], false, false);
                acc2[mi][ni] = __builtin_amdgcn_wmma_f32_16x16x32_bf16(
                    false, fa[mi].v, false, f2[ni].v, (short)0, acc2[mi][ni], false, false);
            }
    };

    int nk = K / 32;              // even
    issue(0);
    for (int i = 0; i < nk - 2; i += 2) {
        issue(1); WAIT_ASYNC_4(); __syncthreads(); compute(0); __syncthreads();
        issue(0); WAIT_ASYNC_4(); __syncthreads(); compute(1); __syncthreads();
    }
    issue(1); WAIT_ASYNC_4(); __syncthreads(); compute(0); __syncthreads();
    WAIT_ASYNC_0(); __syncthreads(); compute(1);

    int lm = 8 * (lane >> 4), ln = lane & 15;
    #pragma unroll
    for (int mi = 0; mi < 2; ++mi)
        #pragma unroll
        for (int ni = 0; ni < 2; ++ni) {
            int col = bn + wn * 32 + ni * 16 + ln;
            float bv1 = bias1[col], bv2 = bias2[col];
            #pragma unroll
            for (int r = 0; r < 8; ++r) {
                int row = bm + wm * 32 + mi * 16 + lm + r;
                float u = acc1[mi][ni][r] + bv1;
                float s = acc2[mi][ni][r] + bv2;
                float h = s * (1.f / (1.f + expf(-s))) * u;
                out[(size_t)row * N + col] = (bf16)h;
            }
        }
}

extern "C" void kernel_launch(void* const* d_in, const int* in_sizes, int n_in,
                              void* d_out, int out_size, void* d_ws, size_t ws_size,
                              hipStream_t stream) {
    const float* x     = (const float*)d_in[0];
    const float* a     = (const float*)d_in[1];
    const float* bvec  = (const float*)d_in[2];
    const float* out_w = (const float*)d_in[3];
    const float* out_b = (const float*)d_in[4];
    const float* n1_w  = (const float*)d_in[5];
    const float* n2_w  = (const float*)d_in[6];
    const float* w1_w  = (const float*)d_in[7];
    const float* w1_b  = (const float*)d_in[8];
    const float* w2_w  = (const float*)d_in[9];
    const float* w2_b  = (const float*)d_in[10];
    const float* w3_w  = (const float*)d_in[11];
    const float* w3_b  = (const float*)d_in[12];

    char* ws = (char*)d_ws;
    const size_t MB = (size_t)1 << 20;
    float* BX   = (float*)ws;                 // 64 MB  (bx -> local scans; later aliased as X1)
    float* X1   = BX;                         // alias (BX consumed before GEMM1 writes X1)
    bf16*  Y    = (bf16*)(ws + 64 * MB);      // 32 MB  (scan output, bf16; later aliased as X2)
    bf16*  X2   = Y;                          // alias (Y consumed by GEMM1 before rms2 writes X2)
    bf16*  Hb   = (bf16*)(ws + 96 * MB);      // 128 MB (gated MLP hidden, bf16)
    bf16*  OWb  = (bf16*)(ws + 224 * MB);     // 2 MB
    bf16*  W1b  = (bf16*)(ws + 226 * MB);     // 8 MB
    bf16*  W2b  = (bf16*)(ws + 234 * MB);     // 8 MB
    bf16*  W3b  = (bf16*)(ws + 242 * MB);     // 8 MB
    float* TAIL = (float*)(ws + 250 * MB);    // 1 MB
    float* S0   = (float*)(ws + 251 * MB);    // 1 MB

    // 1. weight conversion f32 -> bf16
    k_cvt_bf16<<<(D_ * D_) / 256, 256, 0, stream>>>(out_w, OWb, D_ * D_);
    k_cvt_bf16<<<(H_ * D_) / 256, 256, 0, stream>>>(w1_w, W1b, H_ * D_);
    k_cvt_bf16<<<(H_ * D_) / 256, 256, 0, stream>>>(w2_w, W2b, H_ * D_);
    k_cvt_bf16<<<(D_ * H_) / 256, 256, 0, stream>>>(w3_w, W3b, D_ * H_);

    // 2. rmsnorm1 (+ b scaling) -> BX
    k_rms1<<<M_ / 8, 256, 0, stream>>>(x, n1_w, bvec, BX);

    // 3. chunked scan
    k_scan_local<<<(B_ * CH_ * D_) / 256, 256, 0, stream>>>(BX, TAIL, a);
    k_scan_carry<<<(B_ * D_) / 256, 256, 0, stream>>>(TAIL, S0, a);
    k_scan_fix<<<(B_ * CH_ * D_) / 256, 256, 0, stream>>>(BX, S0, a, Y);

    // 4. output projection + residual: X1 = x + Y @ out_w^T + out_b
    k_gemm_resid<<<dim3(D_ / 128, M_ / 128), 256, 0, stream>>>(Y, OWb, out_b, x, X1, D_, D_);

    // 5. rmsnorm2 -> X2 (bf16)
    k_rms2<<<M_ / 8, 256, 0, stream>>>(X1, n2_w, X2);

    // 6. fused dual GEMM + SiLU gate -> Hb
    k_gemm_dual_silu<<<dim3(H_ / 64, M_ / 128), 256, 0, stream>>>(X2, W1b, W2b, w1_b, w2_b, Hb, H_, D_);

    // 7. down projection + residual: out = X1 + Hb @ w3_w^T + w3_b
    k_gemm_resid<<<dim3(D_ / 128, M_ / 128), 256, 0, stream>>>(Hb, W3b, w3_b, X1, (float*)d_out, D_, H_);
}
